// VRNN_45896020525401
// MI455X (gfx1250) — matrix operand order repaired and checked by reading, not compiled, not run on hardware
//
#include <hip/hip_runtime.h>
#include <hip/hip_bf16.h>

typedef _Float16 v8h  __attribute__((ext_vector_type(8)));
typedef _Float16 v16h __attribute__((ext_vector_type(16)));
typedef float    v8f  __attribute__((ext_vector_type(8)));

// Problem dims
static constexpr int Bb = 512, Tt = 256, Xx = 128, Hh = 256, Zz = 64, Mm = 64;
static constexpr int OC = 2 * Xx + 5 * Zz;  // 576 output channels per (b,t)

// f16 weight workspace offsets (in elements)
static constexpr int OFF_WIH = 0;                         // [4H, X+Z] = [1024,192]
static constexpr int OFF_WHH = OFF_WIH + 4 * Hh * (Xx + Zz);
static constexpr int OFF_EW1 = OFF_WHH + 4 * Hh * Hh;     // [64, 384]
static constexpr int OFF_EW2 = OFF_EW1 + Mm * (Hh + Xx);  // [64, 64]
static constexpr int OFF_EMW = OFF_EW2 + Mm * Mm;         // [64, 64]
static constexpr int OFF_EVW = OFF_EMW + Zz * Mm;
static constexpr int OFF_PW1 = OFF_EVW + Zz * Mm;         // [64, 256]
static constexpr int OFF_PW2 = OFF_PW1 + Mm * Hh;
static constexpr int OFF_PMW = OFF_PW2 + Mm * Mm;
static constexpr int OFF_PVW = OFF_PMW + Zz * Mm;
static constexpr int OFF_DW1 = OFF_PVW + Zz * Mm;         // [64, 320]
static constexpr int OFF_DW2 = OFF_DW1 + Mm * (Hh + Zz);  // [128, 64]
static constexpr int OFF_END = OFF_DW2 + Xx * Mm;

// Small (non-LSTM) weights live in LDS: the contiguous block [OFF_EW1, OFF_END)
static constexpr int SWB     = OFF_EW1;            // shared-weight base
static constexpr int SW_ELEM = OFF_END - OFF_EW1;  // 94208 f16 = 184 KB

__global__ void f32_to_f16_kernel(const float* __restrict__ src,
                                  _Float16* __restrict__ dst, int n) {
  int i = blockIdx.x * blockDim.x + threadIdx.x;
  if (i < n) dst[i] = (_Float16)src[i];
}

__device__ __forceinline__ v8f zero8() {
  v8f z = {0.f, 0.f, 0.f, 0.f, 0.f, 0.f, 0.f, 0.f};
  return z;
}

// A fragment (16x32 f16, M x K) from an LDS row-major buffer.
// ISA layout: lanes 0-15 M=lane, elems 0..7 = K k0+0..7, elems 8..15 = K k0+16..23
//             lanes 16-31 same M, K halves +8.
__device__ __forceinline__ v16h frag_a(const _Float16* base, int ld, int k0, int lane) {
  const _Float16* p = base + (lane & 15) * ld + k0 + ((lane >> 4) << 3);
  v8h lo = *(const v8h*)(p);
  v8h hi = *(const v8h*)(p + 16);
  return __builtin_shufflevector(lo, hi, 0, 1, 2, 3, 4, 5, 6, 7,
                                 8, 9, 10, 11, 12, 13, 14, 15);
}

// B fragment (32x16 f16, K x N) for out = inp @ W^T, i.e. B[k][n] = W[n][k].
// Lane holds column n = lane&15, K k0+(lane>=16)*8 + 0..7 and +16..23.
// Works for both global (L2) and LDS-resident row-major W.
__device__ __forceinline__ v16h frag_b(const _Float16* W, int ind,
                                       int n0, int k0, int lane) {
  const _Float16* r = W + (size_t)(n0 + (lane & 15)) * ind + k0 + ((lane >> 4) << 3);
  v8h lo = *(const v8h*)(r);
  v8h hi = *(const v8h*)(r + 16);
  return __builtin_shufflevector(lo, hi, 0, 1, 2, 3, 4, 5, 6, 7,
                                 8, 9, 10, 11, 12, 13, 14, 15);
}

__device__ __forceinline__ v8f wmma16(v16h a, v16h b, v8f c) {
  return __builtin_amdgcn_wmma_f32_16x16x32_f16(false, a, false, b, (short)0, c,
                                                false, false);
}

__device__ __forceinline__ float sigm(float x) { return 1.0f / (1.0f + __expf(-x)); }

__global__ __launch_bounds__(256, 4) void vrnn_scan_kernel(
    const float* __restrict__ x,    // [B,T,X]
    const float* __restrict__ eps,  // [T,B,Z]
    const float* __restrict__ bih, const float* __restrict__ bhh,   // [1024]
    const float* __restrict__ eb1, const float* __restrict__ eb2,   // [64]
    const float* __restrict__ emb, const float* __restrict__ evb,   // [64]
    const float* __restrict__ pb1, const float* __restrict__ pb2,   // [64]
    const float* __restrict__ pmb, const float* __restrict__ pvb,   // [64]
    const float* __restrict__ db1, const float* __restrict__ db2,   // [64],[128]
    const _Float16* __restrict__ wf,  // converted weights (global/L2)
    float* __restrict__ out)          // [B,T,576]
{
  __shared__ alignas(16) _Float16 hbuf[16][Hh];  // h_t (f16)
  __shared__ alignas(16) float    cbuf[16][Hh];  // c_t (f32)
  __shared__ alignas(16) _Float16 xbuf[16][Xx];
  __shared__ alignas(16) _Float16 zbuf[16][Zz];
  __shared__ alignas(16) _Float16 e1[16][Mm];
  __shared__ alignas(16) _Float16 e2[16][Mm];
  __shared__ alignas(16) _Float16 p1[16][Mm];
  __shared__ alignas(16) _Float16 p2[16][Mm];
  __shared__ alignas(16) _Float16 dbuf[16][Mm];
  __shared__ alignas(16) _Float16 wsm[SW_ELEM];  // all non-LSTM weights (184 KB)

  const int tid  = threadIdx.x;
  const int lane = tid & 31;
  const int wv   = tid >> 5;            // wave 0..7
  const int b0   = blockIdx.x * 16;     // batch tile base
  const int mrow = lane & 15;           // N index within a C/D tile
  const int mhi  = (lane >> 4) * 8;     // M offset of element r (M = r + mhi)

  // ---- one-time: pull small weights into LDS (16B vector copies), zero h/c ----
  {
    const uint4* src = (const uint4*)(wf + SWB);
    uint4* dst = (uint4*)&wsm[0];
    for (int i = tid; i < SW_ELEM * 2 / 16; i += 256) dst[i] = src[i];
  }
  for (int i = tid; i < 16 * Hh; i += 256) {
    hbuf[0][i] = (_Float16)0.f;
    cbuf[0][i] = 0.f;
  }

  // ---- hoist all loop-invariant biases into registers ----
  const bool encw = (wv < 4);
  const int  n0s  = (wv & 3) * 16;  // tile base for the small GEMM pipelines
  const float bvL1 = (encw ? eb1 : pb1)[n0s + mrow];
  const float bvL2 = (encw ? eb2 : pb2)[n0s + mrow];
  const float bvM  = (encw ? emb : pmb)[n0s + mrow];
  const float bvV  = (encw ? evb : pvb)[n0s + mrow];
  const float bvD1 = db1[n0s + mrow];       // used by waves 0-3 only
  const float bvD2 = db2[wv * 16 + mrow];   // stage F, all 8 waves
  float bsum[4][2];                          // combined LSTM gate biases
#pragma unroll
  for (int gg = 0; gg < 4; ++gg)
#pragma unroll
    for (int u = 0; u < 2; ++u) {
      int col = wv * 32 + u * 16 + mrow;
      bsum[gg][u] = bih[gg * Hh + col] + bhh[gg * Hh + col];
    }
  __syncthreads();

  const _Float16* sEW1 = wsm + (OFF_EW1 - SWB);
  const _Float16* sEW2 = wsm + (OFF_EW2 - SWB);
  const _Float16* sEMW = wsm + (OFF_EMW - SWB);
  const _Float16* sEVW = wsm + (OFF_EVW - SWB);
  const _Float16* sPW1 = wsm + (OFF_PW1 - SWB);
  const _Float16* sPW2 = wsm + (OFF_PW2 - SWB);
  const _Float16* sPMW = wsm + (OFF_PMW - SWB);
  const _Float16* sPVW = wsm + (OFF_PVW - SWB);
  const _Float16* sDW1 = wsm + (OFF_DW1 - SWB);
  const _Float16* sDW2 = wsm + (OFF_DW2 - SWB);

  for (int t = 0; t < Tt; ++t) {
    // ---- stage the f32 x_t slab into LDS as f16; zero x_logvar output ----
    float* outb = out + ((size_t)b0 * Tt + t) * OC;
    for (int i = tid; i < 16 * Xx; i += 256) {
      int r = i >> 7, c = i & 127;
      xbuf[r][c] = (_Float16)x[((size_t)(b0 + r) * Tt + t) * Xx + c];
      outb[(size_t)r * Tt * OC + Xx + c] = 0.f;  // x_logvar = 0
    }
    // ---- prefetch next step's x (16 rows x 512B) and eps (16 rows x 256B) ----
    if (t + 1 < Tt) {
      if (tid < 64) {  // x[t+1]: 4 x 128B segments per row
        int r = tid >> 2, seg = tid & 3;
        const char* p = (const char*)&x[((size_t)(b0 + r) * Tt + (t + 1)) * Xx];
        __builtin_prefetch(p + seg * 128, 0, 1);
      } else if (tid < 96) {  // eps[t+1]: 2 x 128B segments per row
        int r = (tid - 64) >> 1, seg = (tid - 64) & 1;
        const char* p = (const char*)&eps[((size_t)(t + 1) * Bb + (b0 + r)) * Zz];
        __builtin_prefetch(p + seg * 128, 0, 1);
      }
    }
    __syncthreads();

    // ---- Stage A: enc layer1 (waves 0-3) || prior layer1 (waves 4-7) ----
    if (encw) {
      v8f acc = zero8();
#pragma unroll
      for (int kb = 0; kb < 8; ++kb)  // K 0..255 from h
        acc = wmma16(frag_a(&hbuf[0][0], Hh, kb * 32, lane),
                     frag_b(sEW1, Hh + Xx, n0s, kb * 32, lane), acc);
#pragma unroll
      for (int kb = 0; kb < 4; ++kb)  // K 256..383 from x
        acc = wmma16(frag_a(&xbuf[0][0], Xx, kb * 32, lane),
                     frag_b(sEW1, Hh + Xx, n0s, Hh + kb * 32, lane), acc);
#pragma unroll
      for (int r = 0; r < 8; ++r) {
        float v = acc[r] + bvL1;
        e1[r + mhi][n0s + mrow] = (_Float16)(v > 0.f ? v : 0.f);
      }
    } else {
      v8f acc = zero8();
#pragma unroll
      for (int kb = 0; kb < 8; ++kb)
        acc = wmma16(frag_a(&hbuf[0][0], Hh, kb * 32, lane),
                     frag_b(sPW1, Hh, n0s, kb * 32, lane), acc);
#pragma unroll
      for (int r = 0; r < 8; ++r) {
        float v = acc[r] + bvL1;
        p1[r + mhi][n0s + mrow] = (_Float16)(v > 0.f ? v : 0.f);
      }
    }
    __syncthreads();

    // ---- Stage B: enc layer2 || prior layer2 ----
    if (encw) {
      v8f acc = zero8();
#pragma unroll
      for (int kb = 0; kb < 2; ++kb)
        acc = wmma16(frag_a(&e1[0][0], Mm, kb * 32, lane),
                     frag_b(sEW2, Mm, n0s, kb * 32, lane), acc);
#pragma unroll
      for (int r = 0; r < 8; ++r) {
        float v = acc[r] + bvL2;
        e2[r + mhi][n0s + mrow] = (_Float16)(v > 0.f ? v : 0.f);
      }
    } else {
      v8f acc = zero8();
#pragma unroll
      for (int kb = 0; kb < 2; ++kb)
        acc = wmma16(frag_a(&p1[0][0], Mm, kb * 32, lane),
                     frag_b(sPW2, Mm, n0s, kb * 32, lane), acc);
#pragma unroll
      for (int r = 0; r < 8; ++r) {
        float v = acc[r] + bvL2;
        p2[r + mhi][n0s + mrow] = (_Float16)(v > 0.f ? v : 0.f);
      }
    }
    __syncthreads();

    // ---- Stage C: heads. enc -> mu,logvar,z (waves 0-3) || prior heads (4-7) ----
    if (encw) {
      v8f am = zero8(), av = zero8();
#pragma unroll
      for (int kb = 0; kb < 2; ++kb) {
        v16h a = frag_a(&e2[0][0], Mm, kb * 32, lane);
        am = wmma16(a, frag_b(sEMW, Mm, n0s, kb * 32, lane), am);
        av = wmma16(a, frag_b(sEVW, Mm, n0s, kb * 32, lane), av);
      }
#pragma unroll
      for (int r = 0; r < 8; ++r) {
        int M = r + mhi, col = n0s + mrow;
        float mu = am[r] + bvM;
        float lv = av[r] + bvV;
        float ep = eps[((size_t)t * Bb + (b0 + M)) * Zz + col];
        float z  = mu + ep * __expf(0.5f * lv);
        zbuf[M][col] = (_Float16)z;
        float* orow = outb + (size_t)M * Tt * OC;
        orow[2 * Xx + col]          = z;   // z
        orow[2 * Xx + Zz + col]     = mu;  // mu_enc
        orow[2 * Xx + 2 * Zz + col] = lv;  // logvar_enc
      }
    } else {
      v8f am = zero8(), av = zero8();
#pragma unroll
      for (int kb = 0; kb < 2; ++kb) {
        v16h a = frag_a(&p2[0][0], Mm, kb * 32, lane);
        am = wmma16(a, frag_b(sPMW, Mm, n0s, kb * 32, lane), am);
        av = wmma16(a, frag_b(sPVW, Mm, n0s, kb * 32, lane), av);
      }
#pragma unroll
      for (int r = 0; r < 8; ++r) {
        int M = r + mhi, col = n0s + mrow;
        float* orow = outb + (size_t)M * Tt * OC;
        orow[2 * Xx + 3 * Zz + col] = am[r] + bvM;  // mu_pri
        orow[2 * Xx + 4 * Zz + col] = av[r] + bvV;  // logvar_pri
      }
    }
    __syncthreads();

    // ---- Stage D: LSTM gates (all 8 waves). Wave owns hidden cols [wv*32, wv*32+32) ----
    {
      v8f g[4][2];
#pragma unroll
      for (int gg = 0; gg < 4; ++gg)
#pragma unroll
        for (int u = 0; u < 2; ++u) g[gg][u] = zero8();

      // [x, z] @ Wih^T   (K = 192: 4 blocks x, 2 blocks z) -- B streams from L2
#pragma unroll
      for (int kb = 0; kb < 6; ++kb) {
        v16h a = (kb < 4) ? frag_a(&xbuf[0][0], Xx, kb * 32, lane)
                          : frag_a(&zbuf[0][0], Zz, (kb - 4) * 32, lane);
#pragma unroll
        for (int gg = 0; gg < 4; ++gg)
#pragma unroll
          for (int u = 0; u < 2; ++u) {
            int n0 = gg * Hh + wv * 32 + u * 16;
            g[gg][u] = wmma16(a, frag_b(wf + OFF_WIH, Xx + Zz, n0, kb * 32, lane),
                              g[gg][u]);
          }
      }
      // h @ Whh^T   (K = 256) -- B streams from L2
#pragma unroll
      for (int kb = 0; kb < 8; ++kb) {
        v16h a = frag_a(&hbuf[0][0], Hh, kb * 32, lane);
#pragma unroll
        for (int gg = 0; gg < 4; ++gg)
#pragma unroll
          for (int u = 0; u < 2; ++u) {
            int n0 = gg * Hh + wv * 32 + u * 16;
            g[gg][u] = wmma16(a, frag_b(wf + OFF_WHH, Hh, n0, kb * 32, lane),
                              g[gg][u]);
          }
      }
      __syncthreads();  // everyone done READING hbuf before we overwrite it

#pragma unroll
      for (int u = 0; u < 2; ++u) {
        int col = wv * 32 + u * 16 + mrow;
#pragma unroll
        for (int r = 0; r < 8; ++r) {
          int M = r + mhi;
          float iv = sigm(g[0][u][r] + bsum[0][u]);
          float fv = sigm(g[1][u][r] + bsum[1][u]);
          float gv = tanhf(g[2][u][r] + bsum[2][u]);
          float ov = sigm(g[3][u][r] + bsum[3][u]);
          float cn = fv * cbuf[M][col] + iv * gv;
          cbuf[M][col] = cn;
          hbuf[M][col] = (_Float16)(ov * tanhf(cn));
        }
      }
    }
    __syncthreads();  // h_new visible to all

    // ---- Stage E: dec layer1 (waves 0-3): [h_new, z] @ dec_W1^T ----
    if (encw) {
      v8f acc = zero8();
#pragma unroll
      for (int kb = 0; kb < 8; ++kb)
        acc = wmma16(frag_a(&hbuf[0][0], Hh, kb * 32, lane),
                     frag_b(sDW1, Hh + Zz, n0s, kb * 32, lane), acc);
#pragma unroll
      for (int kb = 0; kb < 2; ++kb)
        acc = wmma16(frag_a(&zbuf[0][0], Zz, kb * 32, lane),
                     frag_b(sDW1, Hh + Zz, n0s, Hh + kb * 32, lane), acc);
#pragma unroll
      for (int r = 0; r < 8; ++r) {
        float v = acc[r] + bvD1;
        dbuf[r + mhi][n0s + mrow] = (_Float16)(v > 0.f ? v : 0.f);
      }
    }
    __syncthreads();

    // ---- Stage F: dec layer2 -> x_mean [16x128], all 8 waves ----
    {
      int n0 = wv * 16;
      v8f acc = zero8();
#pragma unroll
      for (int kb = 0; kb < 2; ++kb)
        acc = wmma16(frag_a(&dbuf[0][0], Mm, kb * 32, lane),
                     frag_b(sDW2, Mm, n0, kb * 32, lane), acc);
#pragma unroll
      for (int r = 0; r < 8; ++r) {
        int M = r + mhi;
        outb[(size_t)M * Tt * OC + n0 + mrow] = acc[r] + bvD2;  // x_mean
      }
    }
    __syncthreads();
  }
}

extern "C" void kernel_launch(void* const* d_in, const int* in_sizes, int n_in,
                              void* d_out, int out_size, void* d_ws, size_t ws_size,
                              hipStream_t stream) {
  _Float16* wf = (_Float16*)d_ws;
  auto cvt = [&](int idx, int off, int n) {
    f32_to_f16_kernel<<<(n + 255) / 256, 256, 0, stream>>>(
        (const float*)d_in[idx], wf + off, n);
  };
  cvt(2,  OFF_WIH, 4 * Hh * (Xx + Zz));  // lstm_Wih
  cvt(4,  OFF_WHH, 4 * Hh * Hh);         // lstm_Whh
  cvt(6,  OFF_EW1, Mm * (Hh + Xx));      // enc_W1
  cvt(8,  OFF_EW2, Mm * Mm);             // enc_W2
  cvt(10, OFF_EMW, Zz * Mm);             // encm_W
  cvt(12, OFF_EVW, Zz * Mm);             // encv_W
  cvt(14, OFF_PW1, Mm * Hh);             // pri_W1
  cvt(16, OFF_PW2, Mm * Mm);             // pri_W2
  cvt(18, OFF_PMW, Zz * Mm);             // prim_W
  cvt(20, OFF_PVW, Zz * Mm);             // priv_W
  cvt(22, OFF_DW1, Mm * (Hh + Zz));      // dec_W1
  cvt(24, OFF_DW2, Xx * Mm);             // dec_W2

  vrnn_scan_kernel<<<Bb / 16, 256, 0, stream>>>(
      (const float*)d_in[0],   // x
      (const float*)d_in[1],   // eps
      (const float*)d_in[3],   // lstm_bih
      (const float*)d_in[5],   // lstm_bhh
      (const float*)d_in[7],   // enc_b1
      (const float*)d_in[9],   // enc_b2
      (const float*)d_in[11],  // encm_b
      (const float*)d_in[13],  // encv_b
      (const float*)d_in[15],  // pri_b1
      (const float*)d_in[17],  // pri_b2
      (const float*)d_in[19],  // prim_b
      (const float*)d_in[21],  // priv_b
      (const float*)d_in[23],  // dec_b1
      (const float*)d_in[25],  // dec_b2
      wf, (float*)d_out);
}
